// SynthesisNetwork_55104430408239
// MI455X (gfx1250) — compile-verified, matrix operand order checked by reading, not run on hardware
//
#include <hip/hip_runtime.h>
#include <hip/hip_bf16.h>

// ---------------------------------------------------------------------------
// SynthesisNetwork layer for MI455X (gfx1250, wave32, WMMA + TDM)
//   styles -> modulated 3x3 conv (WMMA f16 implicit GEMM, A tile via
//   tensor_load_to_lds) -> up-FIR -> leaky-relu -> down-FIR
// ---------------------------------------------------------------------------

typedef __attribute__((ext_vector_type(16))) _Float16 v16h;
typedef __attribute__((ext_vector_type(8)))  _Float16 v8h;
typedef __attribute__((ext_vector_type(8)))  float    v8f;

#if defined(__gfx1250__) && __has_builtin(__builtin_amdgcn_tensor_load_to_lds) && \
    __has_builtin(__builtin_amdgcn_s_wait_tensorcnt)
#define USE_TDM 1
typedef __attribute__((ext_vector_type(4))) unsigned int v4u;
typedef __attribute__((ext_vector_type(8))) int          v8i;
typedef __attribute__((ext_vector_type(4))) int          v4i;
#else
#define USE_TDM 0
#endif

#define NB    4      // batch
#define CIN   512
#define COUT  512
#define WDIM  512
#define HIN   64
#define HP    68     // padded input (pad=2 each side)
#define HO    66     // conv output 64+4-2
#define U1W   138    // width after horizontal up-FIR
#define VR    138    // rows after vertical up-FIR

#define BM 128
#define BN 64        // = 4 output rows x 16 cols
#define BK 32
#define LDA_P 40     // LDS row pitch (halves): 64B data + 16B pad (TDM pad unit)
#define LDB_P 40

// ---- workspace layout (bytes) ----------------------------------------------
static constexpr size_t OFF_SNORM = 0;                                   // 2048 f32
static constexpr size_t OFF_RS    = OFF_SNORM + 2048*4;                  // 2048 f32
static constexpr size_t OFF_W16   = OFF_RS    + 2048*4;                  // 9*512*512 f16
static constexpr size_t OFF_XPAD  = OFF_W16   + (size_t)9*512*512*2;     // 2048*68*68 f16
static constexpr size_t OFF_CONV  = OFF_XPAD  + (size_t)2048*HP*HP*2;    // 2048*66*66 f32
static constexpr size_t OFF_U1    = OFF_CONV  + (size_t)2048*HO*HO*4;    // 2048*66*138 f32
static constexpr size_t OFF_V     = OFF_U1    + (size_t)2048*HO*U1W*4;   // 2048*138*64 f32

// ---------------------------------------------------------------------------
// 1) styles[b,c] = w[b,:] . affine_w[c,:] / sqrt(512) + affine_b[c]
//    snorm = styles * rsqrt(mean(styles^2))        (single block, 512 threads)
// ---------------------------------------------------------------------------
__global__ __launch_bounds__(512)
void styles_kernel(const float* __restrict__ w, const float* __restrict__ aw,
                   const float* __restrict__ ab, float* __restrict__ snorm) {
  __shared__ float red[512];
  const int c = threadIdx.x;
  const float inv = 0.044194173824159216f; // 1/sqrt(512)
  float st[NB];
  float ss = 0.f;
#pragma unroll
  for (int b = 0; b < NB; ++b) {
    float acc = 0.f;
    for (int d = 0; d < WDIM; ++d) acc += w[b*WDIM + d] * aw[c*WDIM + d];
    st[b] = acc * inv + ab[c];
    ss += st[b] * st[b];
  }
  red[c] = ss;
  __syncthreads();
  for (int s = 256; s >= 1; s >>= 1) {
    if (c < s) red[c] += red[c + s];
    __syncthreads();
  }
  const float rs = rsqrtf(red[0] / (float)(NB * CIN));
#pragma unroll
  for (int b = 0; b < NB; ++b) snorm[b*CIN + c] = st[b] * rs;
}

// ---------------------------------------------------------------------------
// 2) rowscale[b,o] = wnorm[o] * rsqrt(wnorm^2 * sum_c snorm^2*sum_k w^2 + eps)
//                  * rsqrt(magnitude_ema)          (one block per o)
// ---------------------------------------------------------------------------
__global__ __launch_bounds__(256)
void rowscale_kernel(const float* __restrict__ cw, const float* __restrict__ snorm,
                     const float* __restrict__ mag, float* __restrict__ rowscale) {
  __shared__ float rT[256];
  __shared__ float rS[NB][256];
  const int o = blockIdx.x;
  const int t = threadIdx.x;
  float pT = 0.f, pS[NB] = {};
  for (int c = t; c < CIN; c += 256) {
    const float* wp = cw + ((size_t)o * CIN + c) * 9;
    float s9 = 0.f;
#pragma unroll
    for (int k = 0; k < 9; ++k) s9 += wp[k] * wp[k];
    pT += s9;
#pragma unroll
    for (int b = 0; b < NB; ++b) {
      float sn = snorm[b*CIN + c];
      pS[b] += s9 * sn * sn;
    }
  }
  rT[t] = pT;
#pragma unroll
  for (int b = 0; b < NB; ++b) rS[b][t] = pS[b];
  __syncthreads();
  for (int s = 128; s >= 1; s >>= 1) {
    if (t < s) {
      rT[t] += rT[t + s];
#pragma unroll
      for (int b = 0; b < NB; ++b) rS[b][t] += rS[b][t + s];
    }
    __syncthreads();
  }
  if (t == 0) {
    const float wn2 = (float)(CIN * 9) / rT[0];  // wnorm^2
    const float wn  = sqrtf(wn2);
    const float ig  = rsqrtf(mag[0]);
#pragma unroll
    for (int b = 0; b < NB; ++b)
      rowscale[b*COUT + o] = wn * rsqrtf(wn2 * rS[b][0] + 1e-8f) * ig;
  }
}

// ---------------------------------------------------------------------------
// 3) f32 conv weight -> f16, layout W16[k][o][c]
// ---------------------------------------------------------------------------
__global__ __launch_bounds__(256)
void wconvert_kernel(const float* __restrict__ cw, _Float16* __restrict__ w16) {
  const int idx = blockIdx.x * 256 + threadIdx.x;     // o*512+c
  const int o = idx >> 9, c = idx & 511;
  const float* wp = cw + (size_t)idx * 9;
#pragma unroll
  for (int k = 0; k < 9; ++k)
    w16[(size_t)k * (COUT*CIN) + (size_t)o * CIN + c] = (_Float16)wp[k];
}

// ---------------------------------------------------------------------------
// 4) xpad[b,c,68,68] = pad2(x[b,c]) * snorm[b,c]  as f16
// ---------------------------------------------------------------------------
__global__ __launch_bounds__(256)
void xpad_kernel(const float* __restrict__ x, const float* __restrict__ snorm,
                 _Float16* __restrict__ xp) {
  const int bc = blockIdx.x;                   // b*512+c
  const float sn = snorm[bc];
  const float* xs = x + (size_t)bc * (HIN*HIN);
  _Float16* xd = xp + (size_t)bc * (HP*HP);
  for (int i = threadIdx.x; i < HP*HP; i += 256) {
    const int yy = i / HP - 2, xx = i % HP - 2;
    float v = 0.f;
    if (yy >= 0 && yy < HIN && xx >= 0 && xx < HIN) v = xs[yy*HIN + xx] * sn;
    xd[i] = (_Float16)v;
  }
}

// ---------------------------------------------------------------------------
// 5) Implicit-GEMM 3x3 conv via V_WMMA_F32_16X16X32_F16.
//    grid = (M tiles=4, N tiles=17y*5x=85, batch=4), 256 threads = 8 waves.
//    Each wave owns a 32x32 output tile (2x2 WMMA tiles).
//    A tile (128x32 f16, row stride 512) is DMA'd by the Tensor Data Mover
//    with LDS padding 16B per 64B row -> padded pitch LDA_P=40 halves.
// ---------------------------------------------------------------------------
__global__ __launch_bounds__(256)
void conv_wmma_kernel(const _Float16* __restrict__ Wk,    // [9][512][512]
                      const _Float16* __restrict__ xpad,  // [B][512][68][68]
                      const float* __restrict__ rowscale, // [B][512]
                      const float* __restrict__ bias,     // [512]
                      float* __restrict__ out) {          // [B][512][66][66]
  __shared__ _Float16 As[BM * LDA_P];   // A tile: [m][k], pitch 40 halves
  __shared__ _Float16 Bs[BN * LDB_P];   // B tile: [n][k]  (stored transposed)

  const int tid  = threadIdx.x;
  const int lane = tid & 31;
  const int wave = tid >> 5;
  const int wm   = wave & 3;            // 0..3 -> M sub-block (32 rows)
  const int wn   = wave >> 2;           // 0..1 -> N sub-block (32 cols)

  const int m0 = blockIdx.x * BM;
  const int nt = blockIdx.y;
  const int x0 = (nt % 5) * 16;         // output-x tile origin
  const int y0 = (nt / 5) * 4;          // output-y tile origin
  const int b  = blockIdx.z;

#if !USE_TDM
  // A loader: thread -> (row, 16-half segment)
  const int ar   = tid >> 1;            // 0..127
  const int aseg = (tid & 1) * 16;      // 0 or 16
#endif
  // B loader: 128 (c,y) pairs, 2 threads/pair, 8 x-values each
  const int pair = tid >> 1;            // 0..127
  const int bcc  = pair & 31;           // c within chunk
  const int byl  = pair >> 5;           // y local 0..3
  const int bx8  = (tid & 1) * 8;

  const size_t xbase = (size_t)b * CIN * (HP*HP);
  const size_t XTOT  = (size_t)NB * CIN * (HP*HP);

  const int lm = lane & 15;             // row/col within 16
  const int kh = (lane >> 4) * 8;       // K-half select per ISA layout

  v8f acc[2][2] = {};

  for (int kk = 0; kk < 9; ++kk) {
    const int ky = kk / 3, kx = kk % 3;
    const _Float16* Wbase = Wk + (size_t)kk * (COUT*CIN);
    for (int c0 = 0; c0 < CIN; c0 += BK) {
      // ---- stage A tile (128 rows x 32 halves) ------------------------
#if USE_TDM
      if (wave == 0) {
        // Tensor DMA descriptor (D#), per CDNA5 ISA ch.8:
        //   2D tile: tile_dim0=32 elem (2B), tile_dim1=128 rows,
        //   tensor_dim0_stride=512, LDS pad: +4 DWORDs every 16 DWORDs.
        const unsigned ldsoff =
            (unsigned)(unsigned long long)(const void*)&As[0];
        const unsigned long long gaddr =
            (unsigned long long)(const void*)(Wbase + (size_t)m0 * CIN + c0);
        v4u g0; v8i g1; v4i gz; v8i gz8;
        g0[0] = 1u;                                   // count=1, user mode
        g0[1] = ldsoff;                               // lds_addr (bytes)
        g0[2] = (unsigned)(gaddr & 0xffffffffu);      // global_addr lo
        g0[3] = (unsigned)((gaddr >> 32) & 0x01ffffffu) | (2u << 30); // type=2
        g1[0] = (1 << 16)            // data_size = 2 bytes
              | (1 << 20)            // pad_enable
              | (3 << 22)            // pad_interval: 2^(3+1)=16 DWORDs (one row)
              | (3 << 25);           // pad_amount: 4 DWORDs = 16 bytes
        g1[1] = (int)(512u << 16);   // tensor_dim0 = 512 (low 16 @ bit48)
        g1[2] = (int)(512u << 16);   // tensor_dim1 = 512 (low 16 @ bit80)
        g1[3] = (int)(32u  << 16);   // tile_dim0 = 32  (@ bit112)
        g1[4] = 128;                 // tile_dim1 = 128 (@ bit128)
        g1[5] = 512;                 // tensor_dim0_stride = 512 (lo32 @ bit160)
        g1[6] = 0;                   // stride0 hi / stride1 lo16
        g1[7] = 4;                   // tensor_dim1_stride = 512*512 (bits 47:16)
        gz[0] = 0; gz[1] = 0; gz[2] = 0; gz[3] = 0;   // 2D: groups 2/3 unused
        gz8[0] = 0; gz8[1] = 0; gz8[2] = 0; gz8[3] = 0;
        gz8[4] = 0; gz8[5] = 0; gz8[6] = 0; gz8[7] = 0;
        __builtin_amdgcn_tensor_load_to_lds(g0, g1, gz, gz, gz8, 0);
      }
#else
      {
        const _Float16* ga = Wbase + (size_t)(m0 + ar) * CIN + c0 + aseg;
        uint4 a0 = *reinterpret_cast<const uint4*>(ga);
        uint4 a1 = *reinterpret_cast<const uint4*>(ga + 8);
        *reinterpret_cast<uint4*>(&As[ar * LDA_P + aseg])     = a0;
        *reinterpret_cast<uint4*>(&As[ar * LDA_P + aseg + 8]) = a1;
        if (c0 + BK < CIN)
          __builtin_prefetch(ga + BK, 0, 0);   // global_prefetch_b8
      }
#endif

      // ---- stage B tile (32c x 4y x 16x), transposed to [n][k] --------
      {
        const int yg = y0 + ky + byl;
        size_t roff = xbase + (size_t)(c0 + bcc) * (HP*HP)
                    + (size_t)yg * HP + (x0 + kx + bx8);
#pragma unroll
        for (int i = 0; i < 8; ++i) {
          size_t idx = roff + i;
          if (idx >= XTOT) idx = XTOT - 1;        // clamp tail overrun
          const int n = byl * 16 + bx8 + i;       // 0..63
          Bs[n * LDB_P + bcc] = xpad[idx];
        }
      }
#if USE_TDM
      if (wave == 0) __builtin_amdgcn_s_wait_tensorcnt(0);  // A tile landed
#endif
      __syncthreads();

      // ---- fragments per ISA 16-bit layouts ---------------------------
      v16h afr[2], bfr[2];
#pragma unroll
      for (int mi = 0; mi < 2; ++mi) {
        const _Float16* ap = &As[(wm*32 + mi*16 + lm) * LDA_P];
        v8h lo = *reinterpret_cast<const v8h*>(ap + kh);
        v8h hi = *reinterpret_cast<const v8h*>(ap + kh + 16);
        afr[mi] = __builtin_shufflevector(lo, hi,
                    0,1,2,3,4,5,6,7,8,9,10,11,12,13,14,15);
      }
#pragma unroll
      for (int ni = 0; ni < 2; ++ni) {
        const _Float16* bp = &Bs[(wn*32 + ni*16 + lm) * LDB_P];
        v8h lo = *reinterpret_cast<const v8h*>(bp + kh);
        v8h hi = *reinterpret_cast<const v8h*>(bp + kh + 16);
        bfr[ni] = __builtin_shufflevector(lo, hi,
                    0,1,2,3,4,5,6,7,8,9,10,11,12,13,14,15);
      }
#pragma unroll
      for (int mi = 0; mi < 2; ++mi)
#pragma unroll
        for (int ni = 0; ni < 2; ++ni)
          acc[mi][ni] = __builtin_amdgcn_wmma_f32_16x16x32_f16(
              false, afr[mi], false, bfr[ni],
              (short)0, acc[mi][ni], false, false);
      __syncthreads();
    }
  }

  // ---- epilogue: scale per output channel, add bias, bounded store ----
#pragma unroll
  for (int mi = 0; mi < 2; ++mi) {
#pragma unroll
    for (int ni = 0; ni < 2; ++ni) {
      const int y = y0 + (wn*2 + ni);
      const int x = x0 + lm;
      if (y < HO && x < HO) {
#pragma unroll
        for (int r = 0; r < 8; ++r) {
          const int m = m0 + wm*32 + mi*16 + r + 8*(lane >> 4);
          const float vv = acc[mi][ni][r] * rowscale[b*COUT + m] + bias[m];
          out[((size_t)(b*COUT + m) * HO + y) * HO + x] = vv;
        }
      }
    }
  }
}

// ---------------------------------------------------------------------------
// 6) Horizontal up-FIR (2x zero-insert + pad 9/8 + 12 taps): only even taps
//    of the zero-inserted signal are live.  conv(66x66) -> u1(66x138)
// ---------------------------------------------------------------------------
__global__ __launch_bounds__(256)
void u1_kernel(const float* __restrict__ conv, const float* __restrict__ fup,
               float* __restrict__ u1) {
  __shared__ float f2[12];
  if (threadIdx.x < 12) f2[threadIdx.x] = 2.0f * fup[11 - threadIdx.x];
  __syncthreads();
  const int bc = blockIdx.x;
  const float* ci = conv + (size_t)bc * (HO*HO);
  float* uo = u1 + (size_t)bc * (HO*U1W);
  for (int idx = threadIdx.x; idx < HO*U1W; idx += 256) {
    const int row = idx / U1W, j = idx % U1W;
    float acc = 0.f;
    const int t0 = (j & 1) ^ 1;                 // (j+t-9) must be even
    for (int t = t0; t < 12; t += 2) {
      const int q = j + t - 9;
      if (q >= 0) {
        const int i = q >> 1;
        if (i < HO) acc += f2[t] * ci[row*HO + i];
      }
    }
    uo[idx] = acc;
  }
}

// ---------------------------------------------------------------------------
// 7) Fused: vertical up-FIR row -> leaky-relu*sqrt(2) -> horizontal down-FIR
//    keeping only even columns.  u1(66x138) -> V(138x64)
// ---------------------------------------------------------------------------
__global__ __launch_bounds__(160)
void u2d1_kernel(const float* __restrict__ u1, const float* __restrict__ fup,
                 const float* __restrict__ fdn, float* __restrict__ v) {
  __shared__ float f2[12], fd[12], row[144];
  const int tid = threadIdx.x;
  if (tid < 12) { f2[tid] = 2.0f * fup[11 - tid]; fd[tid] = fdn[11 - tid]; }
  __syncthreads();
  const int bc = blockIdx.x;
  const int r  = blockIdx.y;                    // 0..137
  const float* ub = u1 + (size_t)bc * (HO*U1W);
  if (tid < U1W) {
    float acc = 0.f;
    const int t0 = (r & 1) ^ 1;
    for (int t = t0; t < 12; t += 2) {
      const int q = r + t - 9;
      if (q >= 0) {
        const int i = q >> 1;
        if (i < HO) acc += f2[t] * ub[i*U1W + tid];
      }
    }
    acc = (acc >= 0.f ? acc : acc * 0.2f) * 1.4142135623730951f;
    row[tid] = acc;
  }
  __syncthreads();
  if (tid < 64) {
    float acc = 0.f;
#pragma unroll
    for (int t = 0; t < 12; ++t) acc += fd[t] * row[2*tid + t];
    v[((size_t)bc * VR + r) * 64 + tid] = acc;
  }
}

// ---------------------------------------------------------------------------
// 8) Vertical down-FIR keeping even rows.  V(138x64) -> out(64x64)
// ---------------------------------------------------------------------------
__global__ __launch_bounds__(64)
void d2_kernel(const float* __restrict__ v, const float* __restrict__ fdn,
               float* __restrict__ out) {
  __shared__ float fd[12];
  if (threadIdx.x < 12) fd[threadIdx.x] = fdn[11 - threadIdx.x];
  __syncthreads();
  const int bc = blockIdx.x;
  const int yo = blockIdx.y;
  const int x  = threadIdx.x;
  const float* vb = v + (size_t)bc * (VR*64);
  float acc = 0.f;
#pragma unroll
  for (int t = 0; t < 12; ++t) acc += fd[t] * vb[(2*yo + t) * 64 + x];
  out[((size_t)bc * 64 + yo) * 64 + x] = acc;
}

// ---------------------------------------------------------------------------
extern "C" void kernel_launch(void* const* d_in, const int* in_sizes, int n_in,
                              void* d_out, int out_size, void* d_ws, size_t ws_size,
                              hipStream_t stream) {
  (void)in_sizes; (void)n_in; (void)out_size; (void)ws_size;
  const float* x   = (const float*)d_in[0];
  const float* w   = (const float*)d_in[1];
  const float* aw  = (const float*)d_in[2];
  const float* ab  = (const float*)d_in[3];
  const float* cw  = (const float*)d_in[4];
  const float* cb  = (const float*)d_in[5];
  const float* fup = (const float*)d_in[6];
  const float* fdn = (const float*)d_in[7];
  const float* mag = (const float*)d_in[8];
  float* out = (float*)d_out;

  char* ws = (char*)d_ws;
  float*    snorm = (float*)(ws + OFF_SNORM);
  float*    rs    = (float*)(ws + OFF_RS);
  _Float16* w16   = (_Float16*)(ws + OFF_W16);
  _Float16* xp    = (_Float16*)(ws + OFF_XPAD);
  float*    conv  = (float*)(ws + OFF_CONV);
  float*    u1    = (float*)(ws + OFF_U1);
  float*    v     = (float*)(ws + OFF_V);

  styles_kernel  <<<1,               512, 0, stream>>>(w, aw, ab, snorm);
  rowscale_kernel<<<COUT,            256, 0, stream>>>(cw, snorm, mag, rs);
  wconvert_kernel<<<(COUT*CIN)/256,  256, 0, stream>>>(cw, w16);
  xpad_kernel    <<<NB*CIN,          256, 0, stream>>>(x, snorm, xp);
  conv_wmma_kernel<<<dim3(COUT/BM, 85, NB), 256, 0, stream>>>(w16, xp, rs, cb, conv);
  u1_kernel      <<<NB*CIN,          256, 0, stream>>>(conv, fup, u1);
  u2d1_kernel    <<<dim3(NB*CIN, VR), 160, 0, stream>>>(u1, fup, fdn, v);
  d2_kernel      <<<dim3(NB*CIN, 64),  64, 0, stream>>>(v, fdn, out);
}
